// AM_Softmax_marginatt_48103633715511
// MI455X (gfx1250) — matrix operand order repaired and checked by reading, not compiled, not run on hardware
//
#include <hip/hip_runtime.h>
#include <hip/hip_bf16.h>
#include <stdint.h>

// ---------------- problem constants ----------------
#define NFEAT    512
#define NCLASSES 85742
#define NDEMOG   4
#define BATCH    512
#define S_SCALE  64.0f
#define LAMBDA_R 100.0f

// ---------------- GEMM tiling ----------------
#define MT 128              // batch rows per workgroup tile
#define NT 128              // classes per tile
#define KT 32               // K step == WMMA K for bf16
#define KSTEPS (NFEAT / KT)                 // 16
#define NTILES ((NCLASSES + NT - 1) / NT)   // 670
#define NCHUNK 64           // workgroups along N per M tile
#define LDSROW 40           // bf16 elems per LDS row: 80 B = 16B aligned, 20 banks -> conflict free

// workspace layout (bytes)
#define OFF_XBF  0u                       // 512*512 bf16      = 524288 B
#define OFF_INVW 524288u                  // 85742 f32         = 342968 B
#define OFF_COSL 867328u                  // 512 f32           = 2048 B
#define OFF_PART 869376u                  // 256*128 f32       = 131072 B  (total ~0.96 MB)

typedef __attribute__((ext_vector_type(16))) __bf16 v16bf;
typedef __attribute__((ext_vector_type(8)))  __bf16 v8bf;
typedef __attribute__((ext_vector_type(8)))  float  v8f;

union U16 { v16bf v; v8bf h[2]; };
union PK4 { __bf16 b[4]; uint2 u; };

// ---------- pass 1a: inverse row norms of W (streams 176 MB once) ----------
__global__ __launch_bounds__(256) void rownorm_inv_kernel(const float* __restrict__ W,
                                                          float* __restrict__ invw) {
  const int lane = threadIdx.x & 31;
  const int wid  = threadIdx.x >> 5;
  const int row  = blockIdx.x * 8 + wid;           // one wave per row
  if (row >= NCLASSES) return;
  const float4* rp = (const float4*)(W + (size_t)row * NFEAT);
  float s = 0.f;
  for (int k = lane; k < NFEAT / 4; k += 32) {
    float4 v = rp[k];
    s += v.x * v.x + v.y * v.y + v.z * v.z + v.w * v.w;
  }
  s += __shfl_xor(s, 16, 32); s += __shfl_xor(s, 8, 32);
  s += __shfl_xor(s, 4, 32);  s += __shfl_xor(s, 2, 32); s += __shfl_xor(s, 1, 32);
  if (lane == 0) invw[row] = 1.0f / fmaxf(sqrtf(s), 1e-12f);
}

// ---------- pass 1b: normalize x rows and convert to bf16 ----------
__global__ __launch_bounds__(256) void xnorm_bf16_kernel(const float* __restrict__ X,
                                                         __bf16* __restrict__ xbf) {
  const int lane = threadIdx.x & 31;
  const int wid  = threadIdx.x >> 5;
  const int row  = blockIdx.x * 8 + wid;           // one wave per row, 512 rows
  if (row >= BATCH) return;
  const float4* rp = (const float4*)(X + (size_t)row * NFEAT);
  float s = 0.f;
  for (int k = lane; k < NFEAT / 4; k += 32) {
    float4 v = rp[k];
    s += v.x * v.x + v.y * v.y + v.z * v.z + v.w * v.w;
  }
  s += __shfl_xor(s, 16, 32); s += __shfl_xor(s, 8, 32);
  s += __shfl_xor(s, 4, 32);  s += __shfl_xor(s, 2, 32); s += __shfl_xor(s, 1, 32);
  const float inv = 1.0f / fmaxf(sqrtf(s), 1e-12f);
  for (int k = lane; k < NFEAT / 4; k += 32) {
    float4 v = rp[k];
    PK4 pk;
    pk.b[0] = (__bf16)(v.x * inv); pk.b[1] = (__bf16)(v.y * inv);
    pk.b[2] = (__bf16)(v.z * inv); pk.b[3] = (__bf16)(v.w * inv);
    *(uint2*)(xbf + (size_t)row * NFEAT + k * 4) = pk.u;
  }
}

// ---------- pass 2: fused bf16-WMMA GEMM + online exp-sum reduction ----------
__global__ __launch_bounds__(256) void gemm_softmax_kernel(const __bf16* __restrict__ xbf,
                                                           const float*  __restrict__ W,
                                                           const float*  __restrict__ invw,
                                                           const long long* __restrict__ label,
                                                           float* __restrict__ coslabel,
                                                           float* __restrict__ partial) {
  // double-buffered tiles: 2 * (10240 + 10240) B = 40 KB LDS (of 320 KB/WGP)
  __shared__ __align__(16) __bf16 XsBuf[2][MT * LDSROW];
  __shared__ __align__(16) __bf16 WsBuf[2][NT * LDSROW];
  __shared__ float Wn[NT];
  __shared__ int   Lab[MT];
  __shared__ float RowSum[MT];

  const int tid   = threadIdx.x;
  const int lane  = tid & 31;
  const int wid   = tid >> 5;          // 8 waves
  const int waveM = wid & 1;           // 2 waves along M (64 rows each)
  const int waveN = wid >> 1;          // 4 waves along N (32 cols each)
  const int h     = lane >> 4;         // half-wave select
  const int l16   = lane & 15;
  const int mt    = blockIdx.y;
  const int mbase = mt * MT;

  // precomputed staging slot coordinates (exact trip counts -> straight-line code)
  const int xs0r = tid >> 2,         xs0p = tid & 3;          // X slot tid
  const int xs1r = (tid + 256) >> 2, xs1p = (tid + 256) & 3;  // X slot tid+256
  const int wr0r = tid >> 3,           wr0p = tid & 7;        // W slots tid + j*256
  const int wr1r = (tid + 256) >> 3,   wr1p = (tid + 256) & 7;
  const int wr2r = (tid + 512) >> 3,   wr2p = (tid + 512) & 7;
  const int wr3r = (tid + 768) >> 3,   wr3p = (tid + 768) & 7;

  if (tid < MT) { RowSum[tid] = 0.f; Lab[tid] = (int)label[mbase + tid]; }
  __syncthreads();

  for (int t = blockIdx.x; t < NTILES; t += NCHUNK) {
    const int cb = t * NT;
    __syncthreads();                               // epilogue of prev tile done
    if (tid < NT) {
      int c = cb + tid;
      Wn[tid] = (c < NCLASSES) ? invw[c] : 0.f;
    }

    v8f acc[4][2];
    const v8f vz = {0.f, 0.f, 0.f, 0.f, 0.f, 0.f, 0.f, 0.f};
    for (int mi = 0; mi < 4; mi++)
      for (int ni = 0; ni < 2; ni++) acc[mi][ni] = vz;

    uint4  xr0, xr1;
    float4 wv0, wv1, wv2, wv3;

    auto fetch = [&](int kk) {
      xr0 = *(const uint4*)(xbf + (size_t)(mbase + xs0r) * NFEAT + kk + xs0p * 8);
      xr1 = *(const uint4*)(xbf + (size_t)(mbase + xs1r) * NFEAT + kk + xs1p * 8);
      const float4 z = make_float4(0.f, 0.f, 0.f, 0.f);
      int c0 = cb + wr0r, c1 = cb + wr1r, c2 = cb + wr2r, c3 = cb + wr3r;
      wv0 = (c0 < NCLASSES) ? *(const float4*)(W + (size_t)c0 * NFEAT + kk + wr0p * 4) : z;
      wv1 = (c1 < NCLASSES) ? *(const float4*)(W + (size_t)c1 * NFEAT + kk + wr1p * 4) : z;
      wv2 = (c2 < NCLASSES) ? *(const float4*)(W + (size_t)c2 * NFEAT + kk + wr2p * 4) : z;
      wv3 = (c3 < NCLASSES) ? *(const float4*)(W + (size_t)c3 * NFEAT + kk + wr3p * 4) : z;
    };
    auto cvt = [](float4 v) -> uint2 {
      PK4 pk;
      pk.b[0] = (__bf16)v.x; pk.b[1] = (__bf16)v.y;
      pk.b[2] = (__bf16)v.z; pk.b[3] = (__bf16)v.w;
      return pk.u;
    };

    fetch(0);                                      // prologue: chunk 0 in flight

    for (int ks = 0; ks < KSTEPS; ks++) {
      __bf16* Xs = XsBuf[ks & 1];
      __bf16* Ws = WsBuf[ks & 1];
      // stage current chunk (straight-line; compiler waits loads here)
      *(uint4*)(Xs + xs0r * LDSROW + xs0p * 8) = xr0;
      *(uint4*)(Xs + xs1r * LDSROW + xs1p * 8) = xr1;
      *(uint2*)(Ws + wr0r * LDSROW + wr0p * 4) = cvt(wv0);
      *(uint2*)(Ws + wr1r * LDSROW + wr1p * 4) = cvt(wv1);
      *(uint2*)(Ws + wr2r * LDSROW + wr2p * 4) = cvt(wv2);
      *(uint2*)(Ws + wr3r * LDSROW + wr3p * 4) = cvt(wv3);
      // issue next chunk's global loads: overlap latency with WMMA below
      if (ks + 1 < KSTEPS) fetch((ks + 1) * KT);
      __syncthreads();

      // A frags: 16x32 bf16, lane<16 -> K{0..7,16..23}, lane>=16 -> K{8..15,24..31}
      v16bf a[4], b[2];
      for (int mi = 0; mi < 4; mi++) {
        int r = waveM * 64 + mi * 16 + l16;
        const __bf16* p0 = Xs + r * LDSROW + h * 8;
        U16 u; u.h[0] = *(const v8bf*)p0; u.h[1] = *(const v8bf*)(p0 + 16);
        a[mi] = u.v;
      }
      // B frags from row-major [class][K]: lane<16 -> K{0..7,16..23}, lane>=16 -> K{8..15,24..31}
      for (int ni = 0; ni < 2; ni++) {
        int r = waveN * 32 + ni * 16 + l16;
        const __bf16* p0 = Ws + r * LDSROW + h * 8;
        U16 u; u.h[0] = *(const v8bf*)p0; u.h[1] = *(const v8bf*)(p0 + 16);
        b[ni] = u.v;
      }
      for (int mi = 0; mi < 4; mi++)
        for (int ni = 0; ni < 2; ni++)
          acc[mi][ni] = __builtin_amdgcn_wmma_f32_16x16x32_bf16(
              false, a[mi], false, b[ni], (short)0, acc[mi][ni], false, false);
      // no trailing barrier: double buffering + next iteration's barrier protect reuse
    }

    // epilogue: scale by 1/||w||, record label cosine, accumulate exp-sums
    const int n0   = cb + waveN * 32 + l16;
    const int n1   = n0 + 16;
    const float iw0 = Wn[waveN * 32 + l16];
    const float iw1 = Wn[waveN * 32 + 16 + l16];
    const bool  v0  = (n0 < NCLASSES);
    const bool  v1  = (n1 < NCLASSES);
    for (int mi = 0; mi < 4; mi++) {
      for (int i = 0; i < 8; i++) {
        int row = waveM * 64 + mi * 16 + i + 8 * h;   // per C-layout (VGPR i -> M=i / i+8)
        float c0 = acc[mi][0][i] * iw0;
        float c1 = acc[mi][1][i] * iw1;
        int lab = Lab[row];
        if (v0 && n0 == lab) coslabel[mbase + row] = c0;
        if (v1 && n1 == lab) coslabel[mbase + row] = c1;
        float e = (v0 ? __expf(S_SCALE * c0) : 0.f) + (v1 ? __expf(S_SCALE * c1) : 0.f);
        // reduce over the 16 lanes of each half-wave (rows differ between halves)
        e += __shfl_xor(e, 1, 32); e += __shfl_xor(e, 2, 32);
        e += __shfl_xor(e, 4, 32); e += __shfl_xor(e, 8, 32);
        if (l16 == 0) atomicAdd(&RowSum[row], e);
      }
    }
  }
  __syncthreads();
  if (tid < MT)
    partial[((size_t)mt * NCHUNK + blockIdx.x) * MT + tid] = RowSum[tid];
}

// ---------- pass 3: combine partials, apply margin, CE mean, loss ----------
__global__ __launch_bounds__(512) void finalize_kernel(const float* __restrict__ partial,
                                                       const float* __restrict__ coslabel,
                                                       const long long* __restrict__ demog,
                                                       const float* __restrict__ margin,
                                                       float* __restrict__ out) {
  __shared__ float red[512];
  const int tid = threadIdx.x;             // one thread per batch row
  const int mt  = tid >> 7;
  const int r   = tid & 127;
  float sum = 0.f;
  for (int c = 0; c < NCHUNK; c++)
    sum += partial[((size_t)mt * NCHUNK + c) * MT + r];
  const float cosl = coslabel[tid];
  const float m    = __expf(margin[(int)demog[tid]]);
  // replace label term exp(S*cosl) with margin-adjusted exp(S*(cosl-m))
  const float adj  = sum - __expf(S_SCALE * cosl) + __expf(S_SCALE * (cosl - m));
  const float ce   = __logf(adj) - S_SCALE * (cosl - m);
  red[tid] = ce;
  __syncthreads();
  for (int s = 256; s > 0; s >>= 1) {
    if (tid < s) red[tid] += red[tid + s];
    __syncthreads();
  }
  if (tid == 0) {
    float tsum = 0.f;
    for (int j = 0; j < NDEMOG; j++) {
      float tj = __expf(margin[j]);
      out[1 + j] = tj;                     // temp
      tsum += tj;
    }
    out[0] = red[0] / (float)BATCH - LAMBDA_R * (tsum / (float)NDEMOG);  // loss
  }
}

extern "C" void kernel_launch(void* const* d_in, const int* in_sizes, int n_in,
                              void* d_out, int out_size, void* d_ws, size_t ws_size,
                              hipStream_t stream) {
  const float*     x      = (const float*)d_in[0];
  const long long* label  = (const long long*)d_in[1];
  const long long* demog  = (const long long*)d_in[2];
  const float*     W      = (const float*)d_in[3];
  const float*     margin = (const float*)d_in[4];
  float*           out    = (float*)d_out;

  char*   ws       = (char*)d_ws;
  __bf16* xbf      = (__bf16*)(ws + OFF_XBF);
  float*  invw     = (float*)(ws + OFF_INVW);
  float*  coslabel = (float*)(ws + OFF_COSL);
  float*  partial  = (float*)(ws + OFF_PART);

  rownorm_inv_kernel<<<(NCLASSES + 7) / 8, 256, 0, stream>>>(W, invw);
  xnorm_bf16_kernel<<<BATCH / 8, 256, 0, stream>>>(x, xbf);
  dim3 grid(NCHUNK, BATCH / MT);
  gemm_softmax_kernel<<<grid, 256, 0, stream>>>(xbf, W, invw, label, coslabel, partial);
  finalize_kernel<<<1, 512, 0, stream>>>(partial, coslabel, demog, margin, out);
}
